// MoEMLP_26972394619540
// MI455X (gfx1250) — compile-verified, matrix operand order checked by reading, not run on hardware
//
#include <hip/hip_runtime.h>
#include <hip/hip_bf16.h>

// ---------------------------------------------------------------------------
// MoE SwiGLU MLP for MI455X (gfx1250), wave32, bf16 WMMA + LDS-staged tiles.
//   x  [T=4096, H=1024] fp32;  Wr [H,9];  Wg/Wu [9,H,I=2816];  Wd [9,I,H]
// Pipeline: zero-out -> f32->bf16 converts -> router/top2 scatter -> prefix ->
//           grouped gate+up GEMM (LDS-staged, WMMA bf16, SiLU*up epilogue) ->
//           grouped down GEMM (LDS-staged, WMMA bf16, atomic f32 combine).
// K-slab = 64 per barrier; next-slab global_prefetch_b8; rcp-based SiLU.
// ---------------------------------------------------------------------------

typedef __attribute__((ext_vector_type(16))) __bf16 v16bf;
typedef __attribute__((ext_vector_type(8)))  float  v8f;
typedef __attribute__((ext_vector_type(4)))  unsigned int v4u;

namespace moe {
constexpr int T = 4096;       // B*S tokens
constexpr int H = 1024;
constexpr int I = 2816;
constexpr int E = 9;
constexpr int MAX_MTILES = T / 16;  // one expert can receive at most T rows
constexpr int KS = 64;              // K-slab per barrier (2 WMMA K-chunks)
constexpr int LDK = 72;             // LDS k-row stride (64 + 8 pad, 16B mult)

union Frag16 {                // 16 bf16 = two 16-byte halves
  v16bf v;
  v4u   q[2];
};

// Per-lane K runs of the 16-bit A/B fragment (ISA 7.12.2):
//   lanes 0-15 : K in [0..7]  and [16..23]
//   lanes 16-31: K in [8..15] and [24..31]
// With LDS staged k-contiguous, a fragment is two ds_load_b128.
__device__ __forceinline__ void load_frag(Frag16& f,
                                          const unsigned short* base,
                                          int half) {
  f.q[0] = *(const v4u*)(base + half * 8);
  f.q[1] = *(const v4u*)(base + 16 + half * 8);
}

// -------------------------------- utility ----------------------------------
__global__ void zero_f32_kernel(float* __restrict__ p, long n) {
  long i = (long)blockIdx.x * blockDim.x + threadIdx.x;
  long stride = (long)gridDim.x * blockDim.x;
  for (; i < n; i += stride) p[i] = 0.0f;
}

__global__ void zero_i32_kernel(int* __restrict__ p, int n) {
  int i = blockIdx.x * blockDim.x + threadIdx.x;
  if (i < n) p[i] = 0;
}

__global__ void f32_to_bf16_kernel(const float* __restrict__ src,
                                   __bf16* __restrict__ dst, long n) {
  long i = (long)blockIdx.x * blockDim.x + threadIdx.x;
  long stride = (long)gridDim.x * blockDim.x;
  for (; i < n; i += stride) dst[i] = (__bf16)src[i];
}

// -------------------------------- router -----------------------------------
__global__ __launch_bounds__(256) void router_kernel(
    const float* __restrict__ x, const float* __restrict__ Wr,
    int* __restrict__ counts, int* __restrict__ tok_list,
    float* __restrict__ wgt_list) {
  int t = blockIdx.x * blockDim.x + threadIdx.x;
  if (t >= T) return;

  float acc[E];
#pragma unroll
  for (int e = 0; e < E; ++e) acc[e] = 0.0f;

  const float* xr = x + (long)t * H;
  for (int h = 0; h < H; ++h) {
    float xv = xr[h];
#pragma unroll
    for (int e = 0; e < E; ++e) acc[e] += xv * Wr[h * E + e];
  }

  float mx = acc[0];
#pragma unroll
  for (int e = 1; e < E; ++e) mx = fmaxf(mx, acc[e]);
  float p[E];
#pragma unroll
  for (int e = 0; e < E; ++e) p[e] = __expf(acc[e] - mx);

  int i1 = 0;
#pragma unroll
  for (int e = 1; e < E; ++e)
    if (p[e] > p[i1]) i1 = e;
  int i2 = (i1 == 0) ? 1 : 0;
#pragma unroll
  for (int e = 0; e < E; ++e)
    if (e != i1 && p[e] > p[i2]) i2 = e;

  float denom = p[i1] + p[i2];
  float w1 = p[i1] / denom;
  float w2 = p[i2] / denom;

  int pos1 = atomicAdd(&counts[i1], 1);
  tok_list[i1 * T + pos1] = t;
  wgt_list[i1 * T + pos1] = w1;
  int pos2 = atomicAdd(&counts[i2], 1);
  tok_list[i2 * T + pos2] = t;
  wgt_list[i2 * T + pos2] = w2;
}

__global__ void prefix_kernel(const int* __restrict__ counts,
                              int* __restrict__ offsets) {
  if (threadIdx.x == 0 && blockIdx.x == 0) {
    int run = 0;
#pragma unroll
    for (int e = 0; e < E; ++e) {
      offsets[e] = run;
      run += counts[e];
    }
    offsets[E] = run;
  }
}

// Stage a 64(K) x 128(N) bf16 weight slab transposed into LDS [n][k].
// Coalesced b64 global reads; b16 LDS scatter stores; prefetch next slab.
__device__ __forceinline__ void stage_b_slab(const __bf16* __restrict__ src,
                                             long rowStride, int k0, int kMax,
                                             int n0,
                                             unsigned short* __restrict__ sh,
                                             int tid) {
  const bool pf = (k0 + KS) < kMax;
#pragma unroll
  for (int i = 0; i < 8; ++i) {
    int g = i * 256 + tid;              // 0..2047
    int kk = g >> 5;                    // 0..63
    int c4 = (g & 31) << 2;             // 0,4,...,124
    const __bf16* p = src + (long)(k0 + kk) * rowStride + n0 + c4;
    uint2 d = *(const uint2*)p;
    if (pf) __builtin_prefetch((const void*)(p + (long)KS * rowStride), 0, 1);
    sh[(c4 + 0) * LDK + kk] = (unsigned short)(d.x & 0xffffu);
    sh[(c4 + 1) * LDK + kk] = (unsigned short)(d.x >> 16);
    sh[(c4 + 2) * LDK + kk] = (unsigned short)(d.y & 0xffffu);
    sh[(c4 + 3) * LDK + kk] = (unsigned short)(d.y >> 16);
  }
}

// --------------------------- gate+up grouped GEMM --------------------------
// grid = (I/128, MAX_MTILES, E); block = 256 (8 waves). Block tile:
// 16 tokens x 128 I-cols; per-wave 16x16 tile, two WMMA chains over K=H.
__global__ __launch_bounds__(256) void gateup_kernel(
    const __bf16* __restrict__ xb, const __bf16* __restrict__ Wg,
    const __bf16* __restrict__ Wu, const int* __restrict__ counts,
    const int* __restrict__ offsets, const int* __restrict__ tok_list,
    __bf16* __restrict__ hidden) {
  __shared__ __align__(16) unsigned short a_sh[16 * LDK];
  __shared__ __align__(16) unsigned short bg_sh[128 * LDK];
  __shared__ __align__(16) unsigned short bu_sh[128 * LDK];

  const int e = blockIdx.z;
  const int cnt = counts[e];
  const int row0 = blockIdx.y * 16;
  if (row0 >= cnt) return;                 // uniform across block

  const int tid = threadIdx.x;
  const int lane = tid & 31;
  const int wave = tid >> 5;
  const int half = lane >> 4;
  const int nloc = wave * 16 + (lane & 15);
  const int n0 = blockIdx.x * 128;
  const long wbase = (long)e * H * I;

  // Hoisted A-staging coords: this thread stages (row mi, 4 bf16 at kd*4).
  const int mi = tid >> 4;                 // 0..15
  const int kd = tid & 15;                 // b64 col 0..15
  const int rowS = row0 + mi;
  const int rowSc = (rowS < cnt) ? rowS : (cnt - 1);
  const long tokS = (long)tok_list[e * T + rowSc];
  const bool rowSok = (rowS < cnt);

  v8f cg = {};
  v8f cu = {};

  for (int k0 = 0; k0 < H; k0 += KS) {
    // stage A (branchless: clamped address + select, one b64 per thread)
    uint2 av = *(const uint2*)(xb + tokS * H + k0 + kd * 4);
    if (!rowSok) { av.x = 0u; av.y = 0u; }
    *(uint2*)(a_sh + mi * LDK + kd * 4) = av;
    // stage Bg / Bu transposed (+ prefetch next slab)
    stage_b_slab(Wg + wbase, I, k0, H, n0, bg_sh, tid);
    stage_b_slab(Wu + wbase, I, k0, H, n0, bu_sh, tid);
    __syncthreads();

#pragma unroll
    for (int c = 0; c < 2; ++c) {          // two 32-K WMMA chunks per slab
      Frag16 a, bg, bu;
      load_frag(a, a_sh + (lane & 15) * LDK + c * 32, half);
      load_frag(bg, bg_sh + nloc * LDK + c * 32, half);
      load_frag(bu, bu_sh + nloc * LDK + c * 32, half);
      cg = __builtin_amdgcn_wmma_f32_16x16x32_bf16(false, a.v, false, bg.v,
                                                   (short)0, cg, false, false);
      cu = __builtin_amdgcn_wmma_f32_16x16x32_bf16(false, a.v, false, bu.v,
                                                   (short)0, cu, false, false);
    }
    __syncthreads();
  }

  const int rowBase = offsets[e] + row0;
  const int col = n0 + nloc;
  const bool full = (row0 + 16 <= cnt);
  if (full) {
#pragma unroll
    for (int r = 0; r < 8; ++r) {
      const int M = (half == 0) ? r : (8 + r);  // C/D layout (ISA 7.12.2)
      float g = cg[r];
      float h = g * __builtin_amdgcn_rcpf(1.0f + __expf(-g)) * cu[r];
      hidden[(long)(rowBase + M) * I + col] = (__bf16)h;
    }
  } else {
#pragma unroll
    for (int r = 0; r < 8; ++r) {
      const int M = (half == 0) ? r : (8 + r);
      if (row0 + M < cnt) {
        float g = cg[r];
        float h = g * __builtin_amdgcn_rcpf(1.0f + __expf(-g)) * cu[r];
        hidden[(long)(rowBase + M) * I + col] = (__bf16)h;
      }
    }
  }
}

// ----------------------------- down grouped GEMM ---------------------------
// grid = (H/128, MAX_MTILES, E). K=I, scale by router weight, atomic combine.
__global__ __launch_bounds__(256) void down_kernel(
    const __bf16* __restrict__ hidden, const __bf16* __restrict__ Wd,
    const int* __restrict__ counts, const int* __restrict__ offsets,
    const int* __restrict__ tok_list, const float* __restrict__ wgt_list,
    float* __restrict__ out) {
  __shared__ __align__(16) unsigned short a_sh[16 * LDK];
  __shared__ __align__(16) unsigned short b_sh[128 * LDK];

  const int e = blockIdx.z;
  const int cnt = counts[e];
  const int row0 = blockIdx.y * 16;
  if (row0 >= cnt) return;                 // uniform across block

  const int tid = threadIdx.x;
  const int lane = tid & 31;
  const int wave = tid >> 5;
  const int half = lane >> 4;
  const int nloc = wave * 16 + (lane & 15);
  const int n0 = blockIdx.x * 128;
  const long wbase = (long)e * I * H;

  const int mi = tid >> 4;
  const int kd = tid & 15;
  const int rowS = row0 + mi;
  const int rowSc = (rowS < cnt) ? rowS : (cnt - 1);
  const long aRowS = (long)(offsets[e] + rowSc);
  const bool rowSok = (rowS < cnt);

  v8f c = {};

  for (int k0 = 0; k0 < I; k0 += KS) {
    uint2 av = *(const uint2*)(hidden + aRowS * I + k0 + kd * 4);
    if (!rowSok) { av.x = 0u; av.y = 0u; }
    *(uint2*)(a_sh + mi * LDK + kd * 4) = av;
    stage_b_slab(Wd + wbase, H, k0, I, n0, b_sh, tid);
    __syncthreads();

#pragma unroll
    for (int cc = 0; cc < 2; ++cc) {
      Frag16 a, b;
      load_frag(a, a_sh + (lane & 15) * LDK + cc * 32, half);
      load_frag(b, b_sh + nloc * LDK + cc * 32, half);
      c = __builtin_amdgcn_wmma_f32_16x16x32_bf16(false, a.v, false, b.v,
                                                  (short)0, c, false, false);
    }
    __syncthreads();
  }

  const int col = n0 + nloc;
  const bool full = (row0 + 16 <= cnt);
  if (full) {
#pragma unroll
    for (int r = 0; r < 8; ++r) {
      const int M = (half == 0) ? r : (8 + r);
      const int row = row0 + M;
      const int t = tok_list[e * T + row];
      const float w = wgt_list[e * T + row];
      atomicAdd(&out[(long)t * H + col], c[r] * w);
    }
  } else {
#pragma unroll
    for (int r = 0; r < 8; ++r) {
      const int M = (half == 0) ? r : (8 + r);
      const int row = row0 + M;
      if (row < cnt) {
        const int t = tok_list[e * T + row];
        const float w = wgt_list[e * T + row];
        atomicAdd(&out[(long)t * H + col], c[r] * w);
      }
    }
  }
}

}  // namespace moe

// ------------------------------- host launcher -----------------------------
extern "C" void kernel_launch(void* const* d_in, const int* in_sizes, int n_in,
                              void* d_out, int out_size, void* d_ws,
                              size_t ws_size, hipStream_t stream) {
  using namespace moe;
  const float* x  = (const float*)d_in[0];
  const float* Wr = (const float*)d_in[1];
  const float* Wg = (const float*)d_in[2];
  const float* Wu = (const float*)d_in[3];
  const float* Wd = (const float*)d_in[4];
  float* out = (float*)d_out;

  // Workspace carve-up (256B aligned). Total ~ 211 MB.
  char* ws = (char*)d_ws;
  size_t off = 0;
  auto take = [&](size_t bytes) -> void* {
    off = (off + 255) & ~(size_t)255;
    void* p = ws + off;
    off += bytes;
    return p;
  };
  __bf16* xb      = (__bf16*)take((size_t)T * H * 2);
  __bf16* Wgb     = (__bf16*)take((size_t)E * H * I * 2);
  __bf16* Wub     = (__bf16*)take((size_t)E * H * I * 2);
  __bf16* Wdb     = (__bf16*)take((size_t)E * I * H * 2);
  __bf16* hidden  = (__bf16*)take((size_t)2 * T * (size_t)I * 2);
  int*    counts  = (int*)take(E * sizeof(int));
  int*    offsets = (int*)take((E + 1) * sizeof(int));
  int*    tok_lst = (int*)take((size_t)E * T * sizeof(int));
  float*  wgt_lst = (float*)take((size_t)E * T * sizeof(float));
  (void)ws_size; (void)in_sizes; (void)n_in; (void)out_size;

  const long nOut = (long)T * H;
  const long nW   = (long)E * H * I;

  // 1) zero output (down_kernel accumulates with atomics) + counters
  zero_f32_kernel<<<1024, 256, 0, stream>>>(out, nOut);
  zero_i32_kernel<<<1, 32, 0, stream>>>(counts, E);

  // 2) fp32 -> bf16 conversions (one streaming pass; bf16 weights ~156MB,
  //    close to L2-resident for the GEMM phase)
  f32_to_bf16_kernel<<<2048, 256, 0, stream>>>(x,  xb,  nOut);
  f32_to_bf16_kernel<<<4096, 256, 0, stream>>>(Wg, Wgb, nW);
  f32_to_bf16_kernel<<<4096, 256, 0, stream>>>(Wu, Wub, nW);
  f32_to_bf16_kernel<<<4096, 256, 0, stream>>>(Wd, Wdb, nW);

  // 3) router: softmax + top-2 + renorm + scatter
  router_kernel<<<T / 256, 256, 0, stream>>>(x, Wr, counts, tok_lst, wgt_lst);
  prefix_kernel<<<1, 32, 0, stream>>>(counts, offsets);

  // 4) grouped gate+up GEMM with fused SiLU*up epilogue
  dim3 gridGU(I / 128, MAX_MTILES, E);
  gateup_kernel<<<gridGU, 256, 0, stream>>>(xb, Wgb, Wub, counts, offsets,
                                            tok_lst, hidden);

  // 5) grouped down GEMM + weighted atomic combine
  dim3 gridDN(H / 128, MAX_MTILES, E);
  down_kernel<<<gridDN, 256, 0, stream>>>(hidden, Wdb, counts, offsets,
                                          tok_lst, wgt_lst, out);
}
